// MambaWrapper_26877905339132
// MI455X (gfx1250) — matrix operand order repaired and checked by reading, not compiled, not run on hardware
//
#include <hip/hip_runtime.h>
#include <hip/hip_bf16.h>
#include <math.h>

// ---------------------------------------------------------------------------
// Bidirectional Mamba block for MI455X (gfx1250, wave32, WMMA + async LDS DMA).
//
// Big GEMMs (in_proj, out_proj): f16 operands pre-converted in workspace,
//   staged to LDS with GLOBAL_LOAD_ASYNC_TO_LDS_B128 (ASYNCcnt), double-
//   buffered, computed with V_WMMA_F32_16X16X32_F16.
// Ragged small GEMMs (x_proj N=80, dt_proj K=48): guarded fp32 path.
// ---------------------------------------------------------------------------

#define D_MODEL 768
#define D_INNER 1536
#define D_STATE 16
#define D_CONV  4
#define DT_RANK 48
#define BATCH   2
#define SEQ     2048
#define NTOK    (BATCH * SEQ)          // 4096 token rows

typedef __attribute__((ext_vector_type(16))) _Float16 v16h;
typedef __attribute__((ext_vector_type(8)))  _Float16 v8h;
typedef __attribute__((ext_vector_type(4)))  _Float16 v4h;
typedef __attribute__((ext_vector_type(8)))  float    v8f;
typedef __attribute__((ext_vector_type(4)))  float    v4f;

__device__ __forceinline__ float silu_f(float x) {
    return x * (1.0f / (1.0f + __expf(-x)));
}
__device__ __forceinline__ float softplus_f(float x) {
    return (x > 20.0f) ? x : log1pf(__expf(x));
}

// ---------------------------------------------------------------------------
// fp32 -> f16 conversion (vectorized, n % 4 == 0 for all call sites)
// ---------------------------------------------------------------------------
__global__ void cvt_f32_f16_kernel(const float* __restrict__ src,
                                   _Float16* __restrict__ dst, int n)
{
    const int i = (blockIdx.x * blockDim.x + threadIdx.x) * 4;
    if (i + 3 < n) {
        const v4f v = *(const v4f*)&src[i];
        v4h h;
#pragma unroll
        for (int q = 0; q < 4; ++q) h[q] = (_Float16)v[q];
        *(v4h*)&dst[i] = h;
    }
}

// ---------------------------------------------------------------------------
// Async-staged WMMA GEMM (f16 inputs): C[M,N] = A[M,K] * W[N,K]^T
// Requires M%64==0, N%64==0, K%32==0, lda/ldw multiples of 8.
// epi = 0: store, 2: C += acc.
// Block = 128 threads (4 waves), block tile 64x64, K chunk 32, double-buffer.
// ---------------------------------------------------------------------------
__global__ __launch_bounds__(128)
void gemm_wmma_f16_async(const _Float16* __restrict__ A, int lda,
                         const _Float16* __restrict__ W, int ldw,
                         float* __restrict__ C, int ldc,
                         int M, int N, int K, int epi)
{
    __shared__ __align__(128) _Float16 As[2][64 * 32];
    __shared__ __align__(128) _Float16 Bs[2][64 * 32];

    const int tid  = threadIdx.x;
    const int lane = tid & 31;
    const int wv   = tid >> 5;

    const int n0 = blockIdx.x * 64;
    const int m0 = blockIdx.y * 64;

    v8f acc[4];
#pragma unroll
    for (int j = 0; j < 4; ++j)
#pragma unroll
        for (int e = 0; e < 8; ++e) acc[j][e] = 0.0f;

    const int khalf = lane >> 4;
    const int l16   = lane & 15;

    // staging coordinates: each lane moves 16-byte chunks (8 halves)
    const int rs = tid >> 2;          // row 0..31 (second iter: +32)
    const int cs = (tid & 3) * 8;     // half-offset {0,8,16,24}

    // issue async copies of one 64x32 K-chunk of A and W into buffer `buf`
    auto stage = [&](int buf, int kc) {
#pragma unroll
        for (int it = 0; it < 2; ++it) {
            const int r = rs + it * 32;
            const unsigned la = (unsigned)(size_t)&As[buf][r * 32 + cs];
            const unsigned lb = (unsigned)(size_t)&Bs[buf][r * 32 + cs];
            const _Float16* ga = &A[(size_t)(m0 + r) * lda + kc + cs];
            const _Float16* gb = &W[(size_t)(n0 + r) * ldw + kc + cs];
            asm volatile("global_load_async_to_lds_b128 %0, %1, off"
                         :: "v"(la), "v"(ga) : "memory");
            asm volatile("global_load_async_to_lds_b128 %0, %1, off"
                         :: "v"(lb), "v"(gb) : "memory");
        }
    };

    stage(0, 0);
    asm volatile("s_wait_asynccnt 0x0" ::: "memory");
    __syncthreads();

    const int kchunks = K / 32;
    for (int kc0 = 0; kc0 < kchunks; ++kc0) {
        const int buf = kc0 & 1;
        if (kc0 + 1 < kchunks) stage(buf ^ 1, (kc0 + 1) * 32);   // overlap DMA

        // A fragment (16x32 f16, contiguous k = 16*khalf + s, same order as B)
        const int arow = (wv * 16 + l16) * 32;
        const v8h a_lo = *(const v8h*)&As[buf][arow + khalf * 16];
        const v8h a_hi = *(const v8h*)&As[buf][arow + khalf * 16 + 8];
        v16h a;
#pragma unroll
        for (int i = 0; i < 8; ++i) { a[i] = a_lo[i]; a[i + 8] = a_hi[i]; }

        // preload all 4 B fragments, then back-to-back WMMAs
        v16h bfrag[4];
#pragma unroll
        for (int j = 0; j < 4; ++j)
            bfrag[j] = *(const v16h*)&Bs[buf][(j * 16 + l16) * 32 + khalf * 16];
#pragma unroll
        for (int j = 0; j < 4; ++j)
            acc[j] = __builtin_amdgcn_wmma_f32_16x16x32_f16(
                false, a, false, bfrag[j], (short)0, acc[j], false, false);

        asm volatile("s_wait_asynccnt 0x0" ::: "memory");
        __syncthreads();
    }

    // epilogue: D layout — lane: N = l16, element r: M = r + 8*khalf
    const int mb = m0 + wv * 16 + khalf * 8;
#pragma unroll
    for (int j = 0; j < 4; ++j) {
        const int gc = n0 + j * 16 + l16;
#pragma unroll
        for (int r = 0; r < 8; ++r) {
            const size_t off = (size_t)(mb + r) * ldc + gc;
            if (epi == 2) C[off] += acc[j][r];
            else          C[off] = acc[j][r];
        }
    }
}

// ---------------------------------------------------------------------------
// Guarded WMMA GEMM (fp32 inputs, ragged shapes):
// C[M,N] = A[M,K(lda)] * W[N,K(ldw)]^T, epi = 0: store, 1: softplus(acc+bias)
// Branch-free staging: clamped index + select.
// ---------------------------------------------------------------------------
__global__ __launch_bounds__(128)
void gemm_wmma_f16_guarded(const float* __restrict__ A, int lda,
                           const float* __restrict__ W, int ldw,
                           float* __restrict__ C, int ldc,
                           int M, int N, int K,
                           const float* __restrict__ bias, int epi)
{
    __shared__ __align__(128) _Float16 As[64 * 32];
    __shared__ __align__(128) _Float16 Bs[64 * 32];

    const int tid  = threadIdx.x;
    const int lane = tid & 31;
    const int wv   = tid >> 5;

    const int n0 = blockIdx.x * 64;
    const int m0 = blockIdx.y * 64;

    v8f acc[4];
#pragma unroll
    for (int j = 0; j < 4; ++j)
#pragma unroll
        for (int e = 0; e < 8; ++e) acc[j][e] = 0.0f;

    const int khalf = lane >> 4;
    const int l16   = lane & 15;

    const int kchunks = (K + 31) / 32;
    for (int kc0 = 0; kc0 < kchunks; ++kc0) {
        const int kc = kc0 * 32;

        for (int e = tid; e < 64 * 32; e += 128) {
            const int r  = e >> 5;
            const int c  = e & 31;
            const int gk = kc + c;
            const int gm = m0 + r;
            const int gn = n0 + r;
            const bool va = (gm < M) && (gk < K);
            const bool vb = (gn < N) && (gk < K);
            const float fa = A[va ? ((size_t)gm * lda + gk) : 0];
            const float fb = W[vb ? ((size_t)gn * ldw + gk) : 0];
            As[e] = (_Float16)(va ? fa : 0.0f);
            Bs[e] = (_Float16)(vb ? fb : 0.0f);
        }
        __syncthreads();

        const int arow = (wv * 16 + l16) * 32;
        const v8h a_lo = *(const v8h*)&As[arow + khalf * 16];
        const v8h a_hi = *(const v8h*)&As[arow + khalf * 16 + 8];
        v16h a;
#pragma unroll
        for (int i = 0; i < 8; ++i) { a[i] = a_lo[i]; a[i + 8] = a_hi[i]; }

        v16h bfrag[4];
#pragma unroll
        for (int j = 0; j < 4; ++j)
            bfrag[j] = *(const v16h*)&Bs[(j * 16 + l16) * 32 + khalf * 16];
#pragma unroll
        for (int j = 0; j < 4; ++j)
            acc[j] = __builtin_amdgcn_wmma_f32_16x16x32_f16(
                false, a, false, bfrag[j], (short)0, acc[j], false, false);
        __syncthreads();
    }

    const int mb = m0 + wv * 16 + khalf * 8;
#pragma unroll
    for (int j = 0; j < 4; ++j) {
        const int gc = n0 + j * 16 + l16;
        if (gc >= N) continue;
#pragma unroll
        for (int r = 0; r < 8; ++r) {
            const int gr = mb + r;
            if (gr >= M) continue;
            float v = acc[j][r];
            const size_t off = (size_t)gr * ldc + gc;
            if (epi == 1) C[off] = softplus_f(v + bias[gc]);
            else          C[off] = v;
        }
    }
}

// ---------------------------------------------------------------------------
// Depthwise causal conv (width 4) + bias + SiLU.
// dir=0: y[l] = silu(b + sum_j w[j]*x[l-3+j]); dir=1: mirrored window.
// x lives in the first D_INNER columns of the xz buffer (stride 2*D_INNER).
// ---------------------------------------------------------------------------
__global__ void conv_silu_kernel(const float* __restrict__ xz,
                                 const float* __restrict__ cw,
                                 const float* __restrict__ cb,
                                 float* __restrict__ xo, int dir)
{
    const int idx = blockIdx.x * blockDim.x + threadIdx.x;
    if (idx >= NTOK * D_INNER) return;
    const int d = idx % D_INNER;
    const int l = (idx / D_INNER) % SEQ;
    const int b = idx / (D_INNER * SEQ);

    float acc = cb[d];
#pragma unroll
    for (int j = 0; j < D_CONV; ++j) {
        const int ls = dir ? (l + (D_CONV - 1) - j) : (l - (D_CONV - 1) + j);
        const bool ok = (ls >= 0) && (ls < SEQ);
        const float xv = xz[ok ? (((size_t)(b * SEQ + ls)) * (2 * D_INNER) + d)
                             : (size_t)d];
        acc += (ok ? xv : 0.0f) * cw[d * D_CONV + j];
    }
    xo[idx] = silu_f(acc);
}

// ---------------------------------------------------------------------------
// Selective scan: one lane per (batch, channel). 16 states in registers.
// dir=0 runs l = 0..SEQ-1, dir=1 runs l = SEQ-1..0 (outputs stay unflipped).
// ---------------------------------------------------------------------------
__global__ __launch_bounds__(256)
void scan_kernel(const float* __restrict__ x,
                 const float* __restrict__ dt,
                 const float* __restrict__ xdbl,
                 const float* __restrict__ A_log,
                 const float* __restrict__ Dp,
                 float* __restrict__ y, int dir)
{
    const int tid = blockIdx.x * blockDim.x + threadIdx.x;
    if (tid >= BATCH * D_INNER) return;
    const int d = tid % D_INNER;
    const int b = tid / D_INNER;

    float Ar[D_STATE];
#pragma unroll
    for (int s = 0; s < D_STATE; ++s)
        Ar[s] = -__expf(A_log[d * D_STATE + s]);

    float hs[D_STATE];
#pragma unroll
    for (int s = 0; s < D_STATE; ++s) hs[s] = 0.0f;

    const float Dd = Dp[d];

    for (int step = 0; step < SEQ; ++step) {
        const int l = dir ? (SEQ - 1 - step) : step;
        const size_t row = (size_t)b * SEQ + l;

        if (step + 1 < SEQ) {
            const int ln = dir ? (SEQ - 2 - step) : (step + 1);
            const size_t rn = (size_t)b * SEQ + ln;
            __builtin_prefetch(&dt[rn * D_INNER + d], 0, 1);
            __builtin_prefetch(&x[rn * D_INNER + d], 0, 1);
            __builtin_prefetch(&xdbl[rn * 80 + DT_RANK], 0, 1);
        }

        const float dtv = dt[row * D_INNER + d];
        const float xv  = x[row * D_INNER + d];
        const float dbx = dtv * xv;
        const float* __restrict__ Bp = &xdbl[row * 80 + DT_RANK];
        const float* __restrict__ Cp = Bp + D_STATE;

        float yv = 0.0f;
#pragma unroll
        for (int s = 0; s < D_STATE; ++s) {
            const float dA = __expf(dtv * Ar[s]);
            hs[s] = dA * hs[s] + dbx * Bp[s];
            yv += hs[s] * Cp[s];
        }
        y[row * D_INNER + d] = yv + xv * Dd;
    }
}

// ---------------------------------------------------------------------------
// y *= silu(z), z = second half of xz.
// ---------------------------------------------------------------------------
__global__ void gate_kernel(float* __restrict__ y, const float* __restrict__ xz)
{
    const int idx = blockIdx.x * blockDim.x + threadIdx.x;
    if (idx >= NTOK * D_INNER) return;
    const int row = idx / D_INNER;
    const int d   = idx % D_INNER;
    const float z = xz[(size_t)row * (2 * D_INNER) + D_INNER + d];
    y[idx] *= silu_f(z);
}

// ---------------------------------------------------------------------------
// Host-side orchestration.
// ---------------------------------------------------------------------------
extern "C" void kernel_launch(void* const* d_in, const int* in_sizes, int n_in,
                              void* d_out, int out_size, void* d_ws, size_t ws_size,
                              hipStream_t stream)
{
    (void)in_sizes; (void)n_in; (void)out_size; (void)ws_size;

    const float* hidden = (const float*)d_in[0];
    float* out = (float*)d_out;

    // workspace partition
    float* ws    = (float*)d_ws;
    float* xz    = ws;                                   // 4096 * 3072 f32
    float* xbuf  = xz   + (size_t)NTOK * 2 * D_INNER;    // 4096 * 1536 f32
    float* xdbl  = xbuf + (size_t)NTOK * D_INNER;        // 4096 * 80   f32
    float* dtb   = xdbl + (size_t)NTOK * 80;             // 4096 * 1536 f32
    float* ybuf  = dtb  + (size_t)NTOK * D_INNER;        // 4096 * 1536 f32
    _Float16* a16 = (_Float16*)(ybuf + (size_t)NTOK * D_INNER);  // 4096*1536 f16
    _Float16* w16 = a16 + (size_t)NTOK * D_INNER;                // 3072*768  f16

    const int ew_threads = 256;
    const int ew_blocks  = (NTOK * D_INNER + ew_threads - 1) / ew_threads;
    auto cvt_blocks = [](int n) { return (n / 4 + 255) / 256; };

    for (int dir = 0; dir < 2; ++dir) {
        const int base = 1 + dir * 9;
        const float* in_proj_w  = (const float*)d_in[base + 0];  // (3072, 768)
        const float* conv_w     = (const float*)d_in[base + 1];  // (1536, 1, 4)
        const float* conv_b     = (const float*)d_in[base + 2];  // (1536,)
        const float* x_proj_w   = (const float*)d_in[base + 3];  // (80, 1536)
        const float* dt_proj_w  = (const float*)d_in[base + 4];  // (1536, 48)
        const float* dt_proj_b  = (const float*)d_in[base + 5];  // (1536,)
        const float* A_log      = (const float*)d_in[base + 6];  // (1536, 16)
        const float* Dv         = (const float*)d_in[base + 7];  // (1536,)
        const float* out_proj_w = (const float*)d_in[base + 8];  // (768, 1536)

        // ---- xz = h @ in_proj_w.T : (4096, 3072), K = 768 [async f16] ----
        cvt_f32_f16_kernel<<<cvt_blocks(NTOK * D_MODEL), 256, 0, stream>>>(
            hidden, a16, NTOK * D_MODEL);
        cvt_f32_f16_kernel<<<cvt_blocks(2 * D_INNER * D_MODEL), 256, 0, stream>>>(
            in_proj_w, w16, 2 * D_INNER * D_MODEL);
        {
            dim3 grid((2 * D_INNER) / 64, NTOK / 64);
            gemm_wmma_f16_async<<<grid, 128, 0, stream>>>(
                a16, D_MODEL, w16, D_MODEL, xz, 2 * D_INNER,
                NTOK, 2 * D_INNER, D_MODEL, 0);
        }

        // ---- depthwise conv + bias + silu -> xbuf ----
        conv_silu_kernel<<<ew_blocks, ew_threads, 0, stream>>>(
            xz, conv_w, conv_b, xbuf, dir);

        // ---- x_dbl = x @ x_proj_w.T : (4096, 80), K = 1536 [guarded] ----
        {
            dim3 grid((80 + 63) / 64, NTOK / 64);
            gemm_wmma_f16_guarded<<<grid, 128, 0, stream>>>(
                xbuf, D_INNER, x_proj_w, D_INNER, xdbl, 80,
                NTOK, 80, D_INNER, nullptr, 0);
        }

        // ---- dt = softplus(x_dbl[:,:48] @ dt_proj_w.T + b) [guarded] ----
        {
            dim3 grid(D_INNER / 64, NTOK / 64);
            gemm_wmma_f16_guarded<<<grid, 128, 0, stream>>>(
                xdbl, 80, dt_proj_w, DT_RANK, dtb, D_INNER,
                NTOK, D_INNER, DT_RANK, dt_proj_b, 1);
        }

        // ---- selective scan -> ybuf ----
        {
            const int threads = 256;
            const int blocks  = (BATCH * D_INNER + threads - 1) / threads;
            scan_kernel<<<blocks, threads, 0, stream>>>(
                xbuf, dtb, xdbl, A_log, Dv, ybuf, dir);
        }

        // ---- gating: ybuf *= silu(z) ----
        gate_kernel<<<ew_blocks, ew_threads, 0, stream>>>(ybuf, xz);

        // ---- out (+)= ybuf @ out_proj_w.T : (4096, 768), K=1536 [async] ----
        cvt_f32_f16_kernel<<<cvt_blocks(NTOK * D_INNER), 256, 0, stream>>>(
            ybuf, a16, NTOK * D_INNER);
        cvt_f32_f16_kernel<<<cvt_blocks(D_MODEL * D_INNER), 256, 0, stream>>>(
            out_proj_w, w16, D_MODEL * D_INNER);
        {
            dim3 grid(D_MODEL / 64, NTOK / 64);
            gemm_wmma_f16_async<<<grid, 128, 0, stream>>>(
                a16, D_INNER, w16, D_INNER, out, D_MODEL,
                NTOK, D_MODEL, D_INNER, dir ? 2 : 0);
        }
    }
}
